// Llama4TextAttention_29892972380383
// MI455X (gfx1250) — matrix operand order repaired and checked by reading, not compiled, not run on hardware
//
#include <hip/hip_runtime.h>
#include <hip/hip_bf16.h>

// ---------------------------------------------------------------------------
// Llama4 attention block on gfx1250 (MI455X).
//   - all GEMM work on v_wmma_f32_16x16x32_bf16 (wave32, fp32 accumulate)
//   - GEMM tiles staged LDS-side via GLOBAL_LOAD_ASYNC_TO_LDS_B128
//     (ASYNCcnt-tracked DMA, double-buffered, s_wait_asynccnt pipelined)
//   - fp32 operands pre-converted to bf16 once (halves GEMM global traffic)
//
// Shapes: B=2, S=2048, HIDDEN=2048, HQ=16, HKV=4, D=128, O3=3072
// Pipeline:
//   k0: convert hidden/qkv_w/o_w -> bf16
//   k1: qkv = Xbf(4096x2048) @ qkv_w_bf^T(3072x2048) -> f32 ws
//   k2: rope(interleaved)+rmsnorm -> q,k bf16 [b][h][s][d]; v^T bf16 [b][h][d][s]
//   k3: causal flash attention, one wave per 16 q-rows, WMMA QK^T and PV
//   k4: out = attn(4096x2048,bf16) @ o_w_bf^T(2048x2048) -> f32 d_out
// ---------------------------------------------------------------------------

typedef __attribute__((ext_vector_type(16))) __bf16       bf16x16;
typedef __attribute__((ext_vector_type(8)))  float        f32x8;
typedef __attribute__((ext_vector_type(4)))  unsigned int u32x4;
typedef __attribute__((ext_vector_type(2)))  unsigned int u32x2;

union FragU { struct { u32x4 lo; u32x4 hi; } u; bf16x16 v; };

// Build a 16-element bf16 fragment from two contiguous 8-half (16B) groups.
static __device__ inline bf16x16 ld_frag(const __bf16* p0, const __bf16* p1) {
  FragU f;
  f.u.lo = *(const u32x4*)p0;
  f.u.hi = *(const u32x4*)p1;
  return f.v;
}

// One wave-level async DMA: 16B per lane, mem -> LDS, tracked by ASYNCcnt.
static __device__ inline void async_b128(unsigned lds_off, const __bf16* gp) {
  asm volatile("global_load_async_to_lds_b128 %0, %1, off"
               :: "v"(lds_off), "v"(gp) : "memory");
}
static __device__ inline void async_b128_o16(unsigned lds_off, const __bf16* gp) {
  // offset:16 applies to BOTH the memory and LDS addresses (ISA 08 §4.4)
  asm volatile("global_load_async_to_lds_b128 %0, %1, off offset:16"
               :: "v"(lds_off), "v"(gp) : "memory");
}

// ---------------------------------------------------------------------------
// fp32 -> bf16 bulk convert (vectorized, n4 = elements/4)
// ---------------------------------------------------------------------------
__global__ __launch_bounds__(256) void f32_to_bf16(
    const float* __restrict__ src, __bf16* __restrict__ dst, int n4) {
  const int i = blockIdx.x * 256 + threadIdx.x;
  if (i < n4) {
    float4 v = ((const float4*)src)[i];
    union { __bf16 h[4]; u32x2 u; } pk;
    pk.h[0] = (__bf16)v.x; pk.h[1] = (__bf16)v.y;
    pk.h[2] = (__bf16)v.z; pk.h[3] = (__bf16)v.w;
    ((u32x2*)dst)[i] = pk.u;
  }
}

// ---------------------------------------------------------------------------
// GEMM: C[m][n] = sum_k A[m][k] * W[n][k]  (A,W bf16 row-major; C fp32)
// Block tile 128x128, BK=32, 256 threads = 8 waves in 4(M) x 2(N) grid,
// each wave 32x64 via 2x4 WMMA tiles. LDS double-buffered; tiles filled by
// GLOBAL_LOAD_ASYNC_TO_LDS_B128 (4 async ops/wave/tile), compute overlaps DMA.
// ---------------------------------------------------------------------------
__global__ __launch_bounds__(256) void gemm_bf16_wmma(
    const __bf16* __restrict__ A, const __bf16* __restrict__ W,
    float* __restrict__ C, int M, int N, int K) {
  constexpr int BK = 32;
  __shared__ __bf16 lA[2][128][BK + 8];   // +8 halfs pad; 16B-aligned chunks
  __shared__ __bf16 lB[2][128][BK + 8];

  const int t    = threadIdx.x;
  const int wave = t >> 5, lane = t & 31, lr = lane & 15, hi = lane >> 4;
  const int wm   = wave >> 1, wn = wave & 1;
  const int bm   = blockIdx.y * 128, bn = blockIdx.x * 128;

  const int lrow = t >> 1;           // 0..127
  const int lseg = (t & 1) * 16;     // 0 or 16 (k offset, halfs)

  f32x8 acc[2][4];
#pragma unroll
  for (int mt = 0; mt < 2; ++mt)
#pragma unroll
    for (int nt = 0; nt < 4; ++nt) acc[mt][nt] = (f32x8)0.0f;

  auto issue_tile = [&](int k0, int buf) {
    const __bf16* ga = A + (size_t)(bm + lrow) * K + k0 + lseg;
    const __bf16* gb = W + (size_t)(bn + lrow) * K + k0 + lseg;
    unsigned la = (unsigned)(uintptr_t)&lA[buf][lrow][lseg];
    unsigned lb = (unsigned)(uintptr_t)&lB[buf][lrow][lseg];
    async_b128(la, ga);
    async_b128_o16(la, ga);
    async_b128(lb, gb);
    async_b128_o16(lb, gb);
  };

  const int niter = K / BK;
  issue_tile(0, 0);

  for (int i = 0; i < niter; ++i) {
    asm volatile("s_wait_asynccnt 0" ::: "memory");  // this wave's tile i in LDS
    __syncthreads();                                 // everyone's tile i in LDS;
                                                     // buf[(i+1)&1] readers done
    if (i + 1 < niter) issue_tile(i + 1, (i + 1) & 1);
    const int buf = i & 1;

    bf16x16 af[2], bfr[4];
#pragma unroll
    for (int mt = 0; mt < 2; ++mt) {
      const int r = wm * 32 + mt * 16 + lr;          // A: lane row M
      af[mt] = ld_frag(&lA[buf][r][hi * 8], &lA[buf][r][16 + hi * 8]);
    }
#pragma unroll
    for (int nt = 0; nt < 4; ++nt) {
      const int r = wn * 64 + nt * 16 + lr;          // B: lane col N
      bfr[nt] = ld_frag(&lB[buf][r][hi * 8], &lB[buf][r][16 + hi * 8]);
    }
#pragma unroll
    for (int mt = 0; mt < 2; ++mt)
#pragma unroll
      for (int nt = 0; nt < 4; ++nt)
        acc[mt][nt] = __builtin_amdgcn_wmma_f32_16x16x32_bf16(
            false, af[mt], false, bfr[nt], (short)0, acc[mt][nt], false, false);
  }

#pragma unroll
  for (int mt = 0; mt < 2; ++mt)
#pragma unroll
    for (int nt = 0; nt < 4; ++nt)
#pragma unroll
      for (int r = 0; r < 8; ++r) {
        const int row = bm + wm * 32 + mt * 16 + r + 8 * hi;
        const int col = bn + wn * 64 + nt * 16 + lr;
        C[(size_t)row * N + col] = acc[mt][nt][r];
      }
}

// ---------------------------------------------------------------------------
// RoPE (interleaved) + RMSNorm. One block per token, 8 waves cover 24 heads
// (16 q + 4 k roped+normed, 4 v converted). Folds 1/sqrt(D) into q.
// ---------------------------------------------------------------------------
__global__ __launch_bounds__(256) void rope_rmsnorm(
    const float* __restrict__ qkv, const float* __restrict__ cosp,
    const float* __restrict__ sinp, const float* __restrict__ nw,
    __bf16* __restrict__ qw, __bf16* __restrict__ kw, __bf16* __restrict__ vtw) {
  const int token = blockIdx.x;            // 0..B*S-1
  const int b = token >> 11, s = token & 2047;
  const int wave = threadIdx.x >> 5, lane = threadIdx.x & 31;

  for (int hh = wave; hh < 24; hh += 8) {
    float4 x = *(const float4*)(qkv + (size_t)token * 3072 + hh * 128 + lane * 4);
    if (hh < 20) {                          // q or k head: rope + rmsnorm
      const int j0 = lane * 2;              // pair indices j0, j0+1
      const float* cr = cosp + (size_t)s * 128;
      const float* sr = sinp + (size_t)s * 128;
      // net effect of deinterleave -> rope -> interleave on original layout:
      float y0 = x.x * cr[j0]      - x.y * sr[j0];
      float y1 = x.y * cr[j0 + 64] + x.x * sr[j0 + 64];
      float y2 = x.z * cr[j0 + 1]  - x.w * sr[j0 + 1];
      float y3 = x.w * cr[j0 + 65] + x.z * sr[j0 + 65];
      float ss = y0 * y0 + y1 * y1 + y2 * y2 + y3 * y3;
#pragma unroll
      for (int m = 1; m < 32; m <<= 1) ss += __shfl_xor(ss, m, 32);
      float rinv = rsqrtf(ss * (1.0f / 128.0f) + 1e-6f);
      if (hh < 16) rinv *= 0.08838834764831845f;   // SCALE = D^-0.5 into q
      const int d = lane * 4;
      y0 *= rinv * nw[d];     y1 *= rinv * nw[d + 1];
      y2 *= rinv * nw[d + 2]; y3 *= rinv * nw[d + 3];
      union { __bf16 h[4]; u32x2 u; } pk;
      pk.h[0] = (__bf16)y0; pk.h[1] = (__bf16)y1;
      pk.h[2] = (__bf16)y2; pk.h[3] = (__bf16)y3;
      __bf16* dst = (hh < 16)
          ? qw + ((size_t)(b * 16 + hh) * 2048 + s) * 128 + d
          : kw + ((size_t)(b * 4 + (hh - 16)) * 2048 + s) * 128 + d;
      *(u32x2*)dst = pk.u;
    } else {                                // v head: convert + transpose store
      const int h = hh - 20, d = lane * 4;
      __bf16* dst = vtw + ((size_t)(b * 4 + h) * 128 + d) * 2048 + s;
      dst[0]        = (__bf16)x.x;
      dst[2048]     = (__bf16)x.y;
      dst[2 * 2048] = (__bf16)x.z;
      dst[3 * 2048] = (__bf16)x.w;
    }
  }
}

// ---------------------------------------------------------------------------
// Causal flash attention. One wave (32 threads) per (b, h, 16 q-rows).
// Q fragments resident in VGPRs; K fragments direct from global (row-major
// [s][d] == B[k=d][n=s] gather); V^T is [d][s] so PV B-fragments are
// contiguous global b128 loads. Online softmax in WMMA D-layout; P
// redistributed via wave-local LDS (s_wait_dscnt guarded).
// ---------------------------------------------------------------------------
__global__ __launch_bounds__(32) void flash_attn(
    const __bf16* __restrict__ q, const __bf16* __restrict__ kk,
    const __bf16* __restrict__ vt, __bf16* __restrict__ o) {
  constexpr int S = 2048, D = 128, HQ = 16, HKV = 4;
  __shared__ __bf16 lP[16][40];

  const int qb = blockIdx.x, h = blockIdx.y, b = blockIdx.z;
  const int kvh = h >> 2;
  const int lane = threadIdx.x, lr = lane & 15, hi = lane >> 4;
  const int m0 = qb * 16;

  const __bf16* qp = q  + ((size_t)(b * HQ + h) * S) * D;
  const __bf16* kp = kk + ((size_t)(b * HKV + kvh) * S) * D;
  const __bf16* vp = vt + ((size_t)(b * HKV + kvh) * D) * S;

  bf16x16 aq[4];
#pragma unroll
  for (int kt = 0; kt < 4; ++kt) {
    const __bf16* r = qp + (size_t)(m0 + lr) * D + kt * 32;
    aq[kt] = ld_frag(r + hi * 8, r + 16 + hi * 8);
  }

  f32x8 acc[8];
#pragma unroll
  for (int nt = 0; nt < 8; ++nt) acc[nt] = (f32x8)0.0f;
  float ms[8], ls[8];
#pragma unroll
  for (int r = 0; r < 8; ++r) { ms[r] = -3.0e38f; ls[r] = 0.0f; }

  const int nsteps = (m0 + 15) / 32 + 1;
  for (int t0 = 0; t0 < nsteps; ++t0) {
    const int kv0 = t0 * 32;
    f32x8 s0 = (f32x8)0.0f, s1 = (f32x8)0.0f;
#pragma unroll
    for (int kt = 0; kt < 4; ++kt) {
      const __bf16* r0 = kp + (size_t)(kv0 + lr) * D      + kt * 32;
      const __bf16* r1 = kp + (size_t)(kv0 + 16 + lr) * D + kt * 32;
      bf16x16 b0 = ld_frag(r0 + hi * 8, r0 + 16 + hi * 8);
      bf16x16 b1 = ld_frag(r1 + hi * 8, r1 + 16 + hi * 8);
      s0 = __builtin_amdgcn_wmma_f32_16x16x32_bf16(false, aq[kt], false, b0, (short)0, s0, false, false);
      s1 = __builtin_amdgcn_wmma_f32_16x16x32_bf16(false, aq[kt], false, b1, (short)0, s1, false, false);
    }
    if (kv0 + 31 > m0) {                      // diagonal tile: causal mask
#pragma unroll
      for (int r = 0; r < 8; ++r) {
        const int row = m0 + r + 8 * hi;
        if (kv0 + lr > row)      s0[r] = -1.0e30f;
        if (kv0 + 16 + lr > row) s1[r] = -1.0e30f;
      }
    }
    float p0[8], p1[8];
#pragma unroll
    for (int r = 0; r < 8; ++r) {
      float v = fmaxf(s0[r], s1[r]);
      v = fmaxf(v, __shfl_xor(v, 1, 32));
      v = fmaxf(v, __shfl_xor(v, 2, 32));
      v = fmaxf(v, __shfl_xor(v, 4, 32));
      v = fmaxf(v, __shfl_xor(v, 8, 32));
      const float mnew  = fmaxf(ms[r], v);
      const float alpha = __expf(ms[r] - mnew);
      ms[r] = mnew;
      p0[r] = __expf(s0[r] - mnew);
      p1[r] = __expf(s1[r] - mnew);
      float rs = p0[r] + p1[r];
      rs += __shfl_xor(rs, 1, 32);
      rs += __shfl_xor(rs, 2, 32);
      rs += __shfl_xor(rs, 4, 32);
      rs += __shfl_xor(rs, 8, 32);
      ls[r] = ls[r] * alpha + rs;
#pragma unroll
      for (int nt = 0; nt < 8; ++nt) acc[nt][r] *= alpha;
      lP[r + 8 * hi][lr]      = (__bf16)p0[r];   // D-layout -> LDS row-major
      lP[r + 8 * hi][16 + lr] = (__bf16)p1[r];
    }
    asm volatile("s_wait_dscnt 0" ::: "memory");
    bf16x16 pa = ld_frag(&lP[lr][hi * 8], &lP[lr][16 + hi * 8]);  // A-layout
    asm volatile("" ::: "memory");             // keep next-iter stores after load
#pragma unroll
    for (int nt = 0; nt < 8; ++nt) {
      const __bf16* r0 = vp + (size_t)(nt * 16 + lr) * S + kv0;
      bf16x16 bv = ld_frag(r0 + hi * 8, r0 + 16 + hi * 8);
      acc[nt] = __builtin_amdgcn_wmma_f32_16x16x32_bf16(false, pa, false, bv, (short)0, acc[nt], false, false);
    }
  }

  // write attn[b][s][h][d] bf16 (token-major for the O-proj GEMM)
#pragma unroll
  for (int nt = 0; nt < 8; ++nt)
#pragma unroll
    for (int r = 0; r < 8; ++r) {
      const int srow = m0 + r + 8 * hi;
      o[(((size_t)b * S + srow) * HQ + h) * D + nt * 16 + lr] =
          (__bf16)(acc[nt][r] / ls[r]);
    }
}

// ---------------------------------------------------------------------------
extern "C" void kernel_launch(void* const* d_in, const int* in_sizes, int n_in,
                              void* d_out, int out_size, void* d_ws, size_t ws_size,
                              hipStream_t stream) {
  (void)in_sizes; (void)n_in; (void)out_size; (void)ws_size;
  const float* hidden = (const float*)d_in[0];
  const float* cosp   = (const float*)d_in[1];
  const float* sinp   = (const float*)d_in[2];
  const float* qkv_w  = (const float*)d_in[3];
  const float* nw     = (const float*)d_in[4];
  const float* o_w    = (const float*)d_in[5];
  float* out = (float*)d_out;

  // workspace layout (88,080,384 B total), liveness-based aliasing:
  //  R0 [0, 50331648)        : qkv f32 (k1 out, k2 in); later attn bf16 (k3/k4)
  //  R1 [50331648, 79691776) : k0/k1: hidden_bf(16.8M)+qkvw_bf(12.6M)
  //                            k2+ : qw(16.8M)+kw(4.2M)+vtw(4.2M)  (same base)
  //  R2 [79691776, 88080384) : ow_bf (live to the end)
  char* ws = (char*)d_ws;
  float*  qkv     = (float*)ws;
  __bf16* attn    = (__bf16*)ws;                      // reuses dead qkv region
  __bf16* hid_bf  = (__bf16*)(ws + 50331648);
  __bf16* qkvw_bf = (__bf16*)(ws + 67108864);
  __bf16* qw      = (__bf16*)(ws + 50331648);         // after k1: hid_bf dead
  __bf16* kw      = (__bf16*)(ws + 67108864);         // after k1: qkvw_bf dead
  __bf16* vtw     = (__bf16*)(ws + 71303168);
  __bf16* ow_bf   = (__bf16*)(ws + 79691776);

  f32_to_bf16<<<8192, 256, 0, stream>>>(hidden, hid_bf, 2097152);   // 4096x2048
  f32_to_bf16<<<6144, 256, 0, stream>>>(qkv_w, qkvw_bf, 1572864);   // 3072x2048
  f32_to_bf16<<<4096, 256, 0, stream>>>(o_w, ow_bf, 1048576);       // 2048x2048

  gemm_bf16_wmma<<<dim3(24, 32), 256, 0, stream>>>(hid_bf, qkvw_bf, qkv, 4096, 3072, 2048);
  rope_rmsnorm<<<4096, 256, 0, stream>>>(qkv, cosp, sinp, nw, qw, kw, vtw);
  flash_attn<<<dim3(128, 16, 2), 32, 0, stream>>>(qw, kw, vtw, attn);
  gemm_bf16_wmma<<<dim3(16, 32), 256, 0, stream>>>(attn, ow_bf, out, 4096, 2048, 2048);
}
